// Quantizer_20753281974680
// MI455X (gfx1250) — matrix operand order repaired and checked by reading, compiled, not run on hardware
//
#include <hip/hip_runtime.h>
#include <hip/hip_bf16.h>

typedef __attribute__((ext_vector_type(16))) _Float16 v16h;
typedef __attribute__((ext_vector_type(8)))  _Float16 v8h;
typedef __attribute__((ext_vector_type(8)))  float    v8f;

#define NBH   32          // b*h
#define W_    64          // windows (= centroids)
#define N_    256         // points per window
#define D_    128         // dim
#define LBH   (W_ * N_)   // 16384 points per (b,h)
#define KPAD  136         // 128 + 8 halves pad (keeps 16B alignment, breaks bank aliasing)

// monotone float -> uint mapping for atomicMax-based argmax
__device__ __forceinline__ unsigned int fkey(float f) {
    unsigned int u = __float_as_uint(f);
    return (u & 0x80000000u) ? ~u : (u | 0x80000000u);
}

// ---------------- init: zero delta_c and best slots ----------------
__global__ void quantizer_init(float* __restrict__ delta, unsigned int* __restrict__ best_u32) {
    int idx = blockIdx.x * 256 + threadIdx.x;           // 262144 threads
    delta[idx] = 0.0f;
    if (idx < NBH * W_ * 2) best_u32[idx] = 0u;         // 2048 u64 slots
}

// ---------------- c_init = l2norm(sum_n x) per window ----------------
__global__ __launch_bounds__(256)
void quantizer_window_mean(const float* __restrict__ x, float* __restrict__ cinit) {
    __shared__ float s[256];
    int tid = threadIdx.x;
    int win = blockIdx.x;                                // (b*8+h)*64+w
    const float* base = x + (size_t)win * N_ * D_;
    int d = tid & 127, half = tid >> 7;
    float acc = 0.0f;
    for (int n = half; n < N_; n += 2) acc += base[(size_t)n * D_ + d];
    s[tid] = acc;
    __syncthreads();
    float sum = 0.0f;
    if (tid < 128) { sum = s[tid] + s[tid + 128]; s[tid] = sum * sum; }
    __syncthreads();
    for (int off = 64; off > 0; off >>= 1) {
        if (tid < off) s[tid] += s[tid + off];
        __syncthreads();
    }
    float rn = rsqrtf(s[0]);
    if (tid < 128) cinit[(size_t)win * D_ + tid] = sum * rn;
}

// ---------------- shared WMMA tile: 64 points x 64 centroids ----------------
// block = 512 threads = 16 waves; wave (pt,ct) computes a 16x16 f32 tile, K=128.
__device__ __forceinline__ void wmma_aff_tile(const float* __restrict__ x,
                                              const float* __restrict__ cmat,
                                              int bh, int chunk,
                                              _Float16 (*xl)[KPAD], _Float16 (*cl)[KPAD],
                                              float (*aff)[65]) {
    int tid = threadIdx.x;
    size_t xbase = ((size_t)bh * LBH + (size_t)chunk * 64) * D_;
    for (int i = tid; i < 64 * D_; i += 512) {
        int r = i >> 7, k = i & 127;
        xl[r][k] = (_Float16)x[xbase + i];
    }
    size_t cbase = (size_t)bh * W_ * D_;
    for (int i = tid; i < W_ * D_; i += 512) {
        int r = i >> 7, k = i & 127;
        cl[r][k] = (_Float16)cmat[cbase + i];
    }
    __syncthreads();

    int lane = tid & 31, wid = tid >> 5;
    int pt = (wid >> 2) * 16;          // point tile base (0..48)
    int ct = (wid & 3) * 16;           // centroid tile base (0..48)
    int row = pt + (lane & 15);
    int col = ct + (lane & 15);
    int ka = (lane & 16) ? 16 : 0;     // A: lanes 16-31 hold K=16..31
    int kb = (lane & 16) ? 16 : 0;     // B: lanes 16-31 hold K=16..31

    v8f acc = {};
#pragma unroll
    for (int k0 = 0; k0 < D_; k0 += 32) {
        v8h alo = *(const v8h*)&xl[row][k0 + ka];
        v8h ahi = *(const v8h*)&xl[row][k0 + ka + 8];
        v8h blo = *(const v8h*)&cl[col][k0 + kb];
        v8h bhi = *(const v8h*)&cl[col][k0 + kb + 8];
        v16h a = __builtin_shufflevector(alo, ahi, 0,1,2,3,4,5,6,7,8,9,10,11,12,13,14,15);
        v16h b = __builtin_shufflevector(blo, bhi, 0,1,2,3,4,5,6,7,8,9,10,11,12,13,14,15);
        acc = __builtin_amdgcn_wmma_f32_16x16x32_f16(false, a, false, b, (short)0, acc, false, false);
    }
    // C layout: lanes 0-15: (M=r, N=lane); lanes 16-31: (M=8+r, N=lane-16)
    int mb = (lane & 16) ? 8 : 0, n = lane & 15;
#pragma unroll
    for (int r = 0; r < 8; r++) aff[pt + mb + r][ct + n] = acc[r];
    __syncthreads();
}

// ---------------- pass 1: aff vs c_init -> assignments + per-centroid argmax ----------------
__global__ __launch_bounds__(512)
void quantizer_aff(const float* __restrict__ x, const float* __restrict__ cinit,
                   int* __restrict__ assign, unsigned long long* __restrict__ best) {
    __shared__ __align__(16) _Float16 xl[64][KPAD];
    __shared__ __align__(16) _Float16 cl[W_][KPAD];
    __shared__ float aff[64][65];
    int tid = threadIdx.x;
    int bh = blockIdx.x >> 8, chunk = blockIdx.x & 255;   // 256 chunks of 64 points
    wmma_aff_tile(x, cinit, bh, chunk, xl, cl, aff);

    if (tid < 64) {
        // per-point argmax over 64 centroids (a_n): first index wins ties (strict >)
        int p = tid, bw = 0;
        float bv = aff[p][0];
        for (int s2 = 1; s2 < W_; s2++) { float v = aff[p][s2]; if (v > bv) { bv = v; bw = s2; } }
        assign[(size_t)bh * LBH + chunk * 64 + p] = bw;
    } else if (tid < 128) {
        // per-centroid block-local argmax over 64 points -> global via u64 atomicMax (a_hw)
        int w = tid - 64, bp = 0;
        float bv = aff[0][w];
        for (int p = 1; p < 64; p++) { float v = aff[p][w]; if (v > bv) { bv = v; bp = p; } }
        unsigned long long key =
            ((unsigned long long)fkey(bv) << 32) | (unsigned int)(chunk * 64 + bp);
        atomicMax(&best[bh * W_ + w], key);
    }
}

// ---------------- pass 2: delta_c accumulation (LDS-binned) ----------------
__global__ __launch_bounds__(256)
void quantizer_accum(const float* __restrict__ x, const int* __restrict__ assign,
                     float* __restrict__ delta) {
    __shared__ float acc[W_ * D_];                        // 32 KB
    int tid = threadIdx.x;
    int bh = blockIdx.x >> 3, chunk = blockIdx.x & 7;     // 8 chunks of 2048 points
    for (int i = tid; i < W_ * D_; i += 256) acc[i] = 0.0f;
    __syncthreads();
    int wid = tid >> 5, lane = tid & 31;
    size_t pbase = (size_t)bh * LBH + (size_t)chunk * 2048;
    for (int lp = wid; lp < 2048; lp += 8) {
        size_t l = pbase + lp;
        int w = assign[l];
        const float4* row = (const float4*)(x + l * (size_t)D_);
        float4 v = row[lane];
        float* ar = acc + w * D_ + lane * 4;
        atomicAdd(ar + 0, v.x); atomicAdd(ar + 1, v.y);
        atomicAdd(ar + 2, v.z); atomicAdd(ar + 3, v.w);
    }
    __syncthreads();
    for (int i = tid; i < W_ * D_; i += 256)
        atomicAdd(&delta[(size_t)bh * W_ * D_ + i], acc[i]);
}

// ---------------- pass 3: centroid update c = l2n(0.5 c + 0.5 l2n(delta + x[best])) ----------------
__global__ __launch_bounds__(128)
void quantizer_update_c(const float* __restrict__ x, const float* __restrict__ cinit,
                        const float* __restrict__ delta, const unsigned long long* __restrict__ best,
                        float* __restrict__ cfinal, float* __restrict__ out_c) {
    __shared__ float s[128];
    int tid = threadIdx.x;
    int r = blockIdx.x;                                   // (bh, w) flat
    int bh = r >> 6;
    unsigned int bl = (unsigned int)best[r];              // winning point index within bh
    float dv = delta[(size_t)r * D_ + tid] + x[((size_t)bh * LBH + bl) * D_ + tid];
    s[tid] = dv * dv;
    __syncthreads();
    for (int off = 64; off > 0; off >>= 1) { if (tid < off) s[tid] += s[tid + off]; __syncthreads(); }
    float ss = s[0];
    __syncthreads();
    float cn = 0.5f * cinit[(size_t)r * D_ + tid] + 0.5f * (dv * rsqrtf(ss));
    s[tid] = cn * cn;
    __syncthreads();
    for (int off = 64; off > 0; off >>= 1) { if (tid < off) s[tid] += s[tid + off]; __syncthreads(); }
    float cf = cn * rsqrtf(s[0]);
    cfinal[(size_t)r * D_ + tid] = cf;
    out_c[(size_t)r * D_ + tid] = cf;
}

// ---------------- pass 4: cos_sim vs c_final -> one-hot ----------------
__global__ __launch_bounds__(512)
void quantizer_assign_out(const float* __restrict__ x, const float* __restrict__ cfinal,
                          float* __restrict__ out) {
    __shared__ __align__(16) _Float16 xl[64][KPAD];
    __shared__ __align__(16) _Float16 cl[W_][KPAD];
    __shared__ float aff[64][65];
    __shared__ int wbest[64];
    int tid = threadIdx.x;
    int bh = blockIdx.x >> 8, chunk = blockIdx.x & 255;
    wmma_aff_tile(x, cfinal, bh, chunk, xl, cl, aff);

    if (tid < 64) {
        int p = tid, bw = 0;
        float bv = aff[p][0];
        for (int s2 = 1; s2 < W_; s2++) { float v = aff[p][s2]; if (v > bv) { bv = v; bw = s2; } }
        wbest[p] = bw;
    }
    __syncthreads();
    size_t obase = ((size_t)bh * LBH + (size_t)chunk * 64) * (size_t)W_;
    for (int i = tid; i < 64 * W_; i += 512) {
        int p = i >> 6, s2 = i & 63;
        out[obase + i] = (s2 == wbest[p]) ? 1.0f : 0.0f;
    }
}

extern "C" void kernel_launch(void* const* d_in, const int* in_sizes, int n_in,
                              void* d_out, int out_size, void* d_ws, size_t ws_size,
                              hipStream_t stream) {
    const float* x = (const float*)d_in[0];
    float* out = (float*)d_out;

    // workspace layout (float offsets)
    float* wsf    = (float*)d_ws;
    float* cinit  = wsf;                                   // 262144 f32
    float* cfinal = wsf + 262144;                          // 262144 f32
    float* delta  = wsf + 524288;                          // 262144 f32
    unsigned long long* best = (unsigned long long*)(wsf + 786432);   // 2048 u64
    int* assign   = (int*)(wsf + 786432 + 4096);           // 524288 i32

    const size_t ONEHOT = (size_t)NBH * LBH * W_;          // 33554432 floats

    quantizer_init<<<1024, 256, 0, stream>>>(delta, (unsigned int*)best);
    quantizer_window_mean<<<NBH * W_, 256, 0, stream>>>(x, cinit);
    quantizer_aff<<<NBH * 256, 512, 0, stream>>>(x, cinit, assign, best);
    quantizer_accum<<<NBH * 8, 256, 0, stream>>>(x, assign, delta);
    quantizer_update_c<<<NBH * W_, 128, 0, stream>>>(x, cinit, delta, best, cfinal, out + ONEHOT);
    quantizer_assign_out<<<NBH * 256, 512, 0, stream>>>(x, cfinal, out);
}
